// LlamaAttentionQKV_54503134986507
// MI455X (gfx1250) — matrix operand order repaired and checked by reading, not compiled, not run on hardware
//
#include <hip/hip_runtime.h>

typedef __attribute__((ext_vector_type(16))) _Float16 v16h;
typedef __attribute__((ext_vector_type(8)))  _Float16 v8h;
typedef __attribute__((ext_vector_type(8)))  float    v8f;
typedef __attribute__((ext_vector_type(4)))  float    v4f;

namespace {
constexpr int kB    = 2;
constexpr int kS    = 2048;
constexpr int kHQ   = 32;
constexpr int kHKV  = 8;
constexpr int kD    = 128;
constexpr int kGrp  = kHQ / kHKV;   // 4
constexpr int kQTile = 64;          // queries per workgroup (4 waves x 16 rows)
constexpr int kKBlk  = 32;          // keys per inner iteration (K-dim of PV wmma)
constexpr int kKPad  = 136;         // 128 + 8 halfs  (272B rows, 16B aligned)
constexpr int kVPad  = 40;          // 32 + 8 halfs   (80B rows, 16B aligned)
constexpr float kNegBig = -1.0e30f;
}

__device__ __forceinline__ v8f wmma_f16(v16h a, v16h b, v8f c) {
  return __builtin_amdgcn_wmma_f32_16x16x32_f16(false, a, false, b, (short)0, c,
                                                false, false);
}

// reductions across the 16 lanes sharing a score row (xor<=8 stays in half-wave)
__device__ __forceinline__ float row_max16(float t) {
  t = fmaxf(t, __shfl_xor(t, 1, 32));
  t = fmaxf(t, __shfl_xor(t, 2, 32));
  t = fmaxf(t, __shfl_xor(t, 4, 32));
  t = fmaxf(t, __shfl_xor(t, 8, 32));
  return t;
}
__device__ __forceinline__ float row_sum16(float t) {
  t += __shfl_xor(t, 1, 32);
  t += __shfl_xor(t, 2, 32);
  t += __shfl_xor(t, 4, 32);
  t += __shfl_xor(t, 8, 32);
  return t;
}

__global__ __launch_bounds__(128) void
llama_attn_fwd(const float* __restrict__ Qg, const float* __restrict__ Kg,
               const float* __restrict__ Vg, const float* __restrict__ Cg,
               const float* __restrict__ Sg, const float* __restrict__ scale_p,
               float* __restrict__ Og)
{
  // cooperative staging buffers (shared by the 4 waves) + per-wave P scratch
  __shared__ __align__(16) _Float16 kld[kKBlk][kKPad];   // RoPE'd K, f16 [key][dim]
  __shared__ __align__(16) _Float16 vld[kD][kVPad];      // V transposed, f16 [dim][key]
  __shared__ __align__(16) _Float16 psc[4][16][40];      // per-wave P transpose

  const int lane = threadIdx.x & 31;
  const int wave = threadIdx.x >> 5;
  const int hi   = lane >> 4;        // half-wave select
  const int ln   = lane & 15;
  const int qt = blockIdx.x;
  const int h  = blockIdx.y;
  const int b  = blockIdx.z;
  const int hk = h / kGrp;
  const int q0 = qt * kQTile + wave * 16;
  const float scale = scale_p[0];

  // staging split: 4 threads per key row, each covering paired dim groups
  const int srow = threadIdx.x >> 2;  // key row within block (0..31)
  const int ssub = threadIdx.x & 3;

  const size_t kvStride = (size_t)kHKV * kD;                       // 1024 floats
  const float* kb0 = Kg + (size_t)b * kS * kvStride + (size_t)hk * kD;
  const float* vb0 = Vg + (size_t)b * kS * kvStride + (size_t)hk * kD;

  // ---- Q tile: row = q0+ln for both half-waves; lane half picks dim subsets.
  const float* qp = Qg + (((size_t)b * kS + (size_t)(q0 + ln)) * kHQ + h) * kD;
  const float* cq = Cg + (size_t)(q0 + ln) * kD;
  const float* sq = Sg + (size_t)(q0 + ln) * kD;

  float qraw[4][16];
#pragma unroll
  for (int c = 0; c < 4; ++c) {
#pragma unroll
    for (int g = 0; g < 2; ++g) {
      const int base = 32 * c + 16 * g + 8 * hi;
      v4f a0 = *(const v4f*)(qp + base);
      v4f a1 = *(const v4f*)(qp + base + 4);
#pragma unroll
      for (int j = 0; j < 4; ++j) {
        qraw[c][8 * g + j]     = a0[j];
        qraw[c][8 * g + 4 + j] = a1[j];
      }
    }
  }
  // RoPE across chunk pairs (c, c+2) == dims (d, d+64); pre-apply softmax scale.
  v16h aq[4];
#pragma unroll
  for (int c = 0; c < 2; ++c) {
#pragma unroll
    for (int e = 0; e < 16; ++e) {
      const int d = 32 * c + 16 * (e >> 3) + 8 * hi + (e & 7);   // d < 64
      const float cd = cq[d], sd = sq[d];
      const float x = qraw[c][e], y = qraw[c + 2][e];
      aq[c][e]     = (_Float16)((x * cd - y * sd) * scale);
      aq[c + 2][e] = (_Float16)((y * cd + x * sd) * scale);
    }
  }

  // ---- online softmax state (per lane: 8 rows of this half-wave's group)
  v8f o[8];
  float m[8], l[8];
#pragma unroll
  for (int r = 0; r < 8; ++r) {
    m[r] = -3.0e38f;
    l[r] = 0.0f;
#pragma unroll
    for (int n = 0; n < 8; ++n) o[n][r] = 0.0f;
  }

  const int nblk_w   = ((q0 + 15) >> 5) + 1;   // this wave's causal block count
  const int nblk_max = 2 * qt + 2;             // block count of last wave (uniform)

  for (int bi = 0; bi < nblk_max; ++bi) {
    const int n0 = bi * kKBlk;

    __syncthreads();   // previous iteration's consumers done before overwrite

    // ---- cooperative staging: K (RoPE'd, f16) and V (transposed, f16) to LDS
    {
      const int j = n0 + srow;
      const float* kp = kb0 + (size_t)j * kvStride;
      const float* ck = Cg + (size_t)j * kD;
      const float* sk = Sg + (size_t)j * kD;
      const int d0 = 16 * ssub;                 // dims [d0,d0+16) + [d0+64,d0+80)
      float x[16], y[16], cs[16], sn[16];
#pragma unroll
      for (int t = 0; t < 4; ++t) {
        *(v4f*)(x + 4 * t)  = *(const v4f*)(kp + d0 + 4 * t);
        *(v4f*)(y + 4 * t)  = *(const v4f*)(kp + d0 + 64 + 4 * t);
        *(v4f*)(cs + 4 * t) = *(const v4f*)(ck + d0 + 4 * t);
        *(v4f*)(sn + 4 * t) = *(const v4f*)(sk + d0 + 4 * t);
      }
      v8h w0, w1, w2, w3;
#pragma unroll
      for (int e = 0; e < 8; ++e) {
        w0[e] = (_Float16)(x[e] * cs[e] - y[e] * sn[e]);
        w1[e] = (_Float16)(x[8 + e] * cs[8 + e] - y[8 + e] * sn[8 + e]);
        w2[e] = (_Float16)(y[e] * cs[e] + x[e] * sn[e]);
        w3[e] = (_Float16)(y[8 + e] * cs[8 + e] + x[8 + e] * sn[8 + e]);
      }
      *(v8h*)(&kld[srow][d0])      = w0;
      *(v8h*)(&kld[srow][d0 + 8])  = w1;
      *(v8h*)(&kld[srow][d0 + 64]) = w2;
      *(v8h*)(&kld[srow][d0 + 72]) = w3;

      // V: coalesced f32 read, f16 transpose-scatter into [dim][key]
      const float* vp = vb0 + (size_t)j * kvStride;
      const int vd0 = 32 * ssub;
#pragma unroll
      for (int t = 0; t < 8; ++t) {
        v4f w = *(const v4f*)(vp + vd0 + 4 * t);
#pragma unroll
        for (int u = 0; u < 4; ++u)
          vld[vd0 + 4 * t + u][srow] = (_Float16)w[u];
      }

      if (bi + 1 < nblk_max) {                  // hint next block's cachelines
        const float* kn = kb0 + (size_t)(j + kKBlk) * kvStride + d0;
        const float* vn = vb0 + (size_t)(j + kKBlk) * kvStride + vd0;
        __builtin_prefetch(kn, 0, 0);
        __builtin_prefetch(vn, 0, 0);
      }
    }
    __syncthreads();   // staged tiles visible to all waves

    if (bi >= nblk_w) continue;                 // past this wave's causal limit
    const bool need_mask = (n0 + kKBlk - 1) > q0;

    // ---- scores: S = Q * K^T (4 K-dim chunks x 2 key-column halves)
    v8f s0, s1;
#pragma unroll
    for (int r = 0; r < 8; ++r) { s0[r] = 0.0f; s1[r] = 0.0f; }
#pragma unroll
    for (int which = 0; which < 2; ++which) {
#pragma unroll
      for (int c = 0; c < 4; ++c) {
        const _Float16* kr = &kld[16 * which + ln][32 * c + 8 * hi];
        v8h klo = *(const v8h*)(kr);
        v8h khi = *(const v8h*)(kr + 16);
        v16h bk;
#pragma unroll
        for (int e = 0; e < 8; ++e) { bk[e] = klo[e]; bk[8 + e] = khi[e]; }
        if (which == 0) s0 = wmma_f16(aq[c], bk, s0);
        else            s1 = wmma_f16(aq[c], bk, s1);
      }
    }

    // ---- causal mask + online softmax update (C layout: row = r + 8*hi)
#pragma unroll
    for (int r = 0; r < 8; ++r) {
      const int row = q0 + r + 8 * hi;
      if (need_mask) {
        if (n0 + ln > row)      s0[r] = kNegBig;
        if (n0 + 16 + ln > row) s1[r] = kNegBig;
      }
      const float rmax  = row_max16(fmaxf(s0[r], s1[r]));
      const float mnew  = fmaxf(m[r], rmax);
      const float alpha = __expf(m[r] - mnew);
      m[r] = mnew;
      const float p0 = __expf(s0[r] - mnew);
      const float p1 = __expf(s1[r] - mnew);
      s0[r] = p0;
      s1[r] = p1;
      l[r] = l[r] * alpha + row_sum16(p0 + p1);
#pragma unroll
      for (int n = 0; n < 8; ++n) o[n][r] *= alpha;
    }

    // ---- C-layout -> A-layout transpose of P through per-wave LDS
    _Float16 (*ps)[40] = psc[wave];
#pragma unroll
    for (int r = 0; r < 8; ++r) {
      ps[r + 8 * hi][ln]      = (_Float16)s0[r];
      ps[r + 8 * hi][16 + ln] = (_Float16)s1[r];
    }
    // same-wave LDS is in-order; compiler inserts s_wait_dscnt on the reads
    v8h plo = *(const v8h*)(&ps[ln][8 * hi]);
    v8h phi = *(const v8h*)(&ps[ln][16 + 8 * hi]);
    v16h ap;
#pragma unroll
    for (int e = 0; e < 8; ++e) { ap[e] = plo[e]; ap[8 + e] = phi[e]; }

    // ---- O += P * V : B-fragment = two contiguous ds_load_b128 from vld
#pragma unroll
    for (int n = 0; n < 8; ++n) {
      const _Float16* vr = &vld[16 * n + ln][8 * hi];
      v8h vlo = *(const v8h*)(vr);
      v8h vhi = *(const v8h*)(vr + 16);
      v16h bv;
#pragma unroll
      for (int e = 0; e < 8; ++e) { bv[e] = vlo[e]; bv[8 + e] = vhi[e]; }
      o[n] = wmma_f16(ap, bv, o[n]);
    }
  }

  // ---- epilogue: normalize and store fp32 output [B,S,HQ,D]
#pragma unroll
  for (int r = 0; r < 8; ++r) {
    const int row = q0 + r + 8 * hi;
    const float inv = 1.0f / l[r];
    float* op = Og + (((size_t)b * kS + row) * kHQ + h) * kD;
#pragma unroll
    for (int n = 0; n < 8; ++n) op[16 * n + ln] = o[n][r] * inv;
  }
}

extern "C" void kernel_launch(void* const* d_in, const int* in_sizes, int n_in,
                              void* d_out, int out_size, void* d_ws, size_t ws_size,
                              hipStream_t stream) {
  (void)in_sizes; (void)n_in; (void)out_size; (void)d_ws; (void)ws_size;
  const float* Q  = (const float*)d_in[0];
  const float* K  = (const float*)d_in[1];
  const float* V  = (const float*)d_in[2];
  const float* C  = (const float*)d_in[3];
  const float* S  = (const float*)d_in[4];
  // d_in[5] = attention_mask: pure causal -> regenerated in-kernel
  const float* sc = (const float*)d_in[6];
  float* O = (float*)d_out;

  dim3 grid(kS / kQTile, kHQ, kB);   // (32, 32, 2)
  dim3 block(128);                   // 4 waves, one 16-row WMMA tile each
  hipLaunchKernelGGL(llama_attn_fwd, grid, block, 0, stream,
                     Q, K, V, C, S, sc, O);
}
